// ElectronicSchNet_38362647888621
// MI455X (gfx1250) — matrix operand order repaired and checked by reading, not compiled, not run on hardware
//
#include <hip/hip_runtime.h>
#include <hip/hip_bf16.h>

// Sizes from the reference
#define N_TOT 32
#define MM    4
#define DD    128
#define KK    64
#define FF    32
#define LL    3
#define BB    1024
#define HWID  45
#define HP    48   // H padded for GEMM1 output (N dim)
#define LOG2F_ 0.69314718056f

typedef __attribute__((ext_vector_type(16))) _Float16 v16h;
typedef __attribute__((ext_vector_type(8)))  _Float16 v8h;
typedef __attribute__((ext_vector_type(8)))  float    v8f;

// ---- f16 weight workspace layout (offsets in halves) ----
#define W1_SZ  (32 * 48)    // per (l,c): [f=32][hpad=48]
#define W2_SZ  (64 * 64)    // per (l,c): [hpad=64][k=64]
#define HWC_SZ (128 * 64)   // per l:     [d=128][k=64]
#define GW_SZ  (64 * 128)   // per (l,c): [k=64][d=128]
#define W1_OFF  0
#define W2_OFF  (W1_OFF + 9 * W1_SZ)
#define HWC_OFF (W2_OFF + 9 * W2_SZ)
#define GW_OFF  (HWC_OFF + 3 * HWC_SZ)

#define BUF_ELEMS (BB * N_TOT * KK)   // 2097152 f32 per z/h buffer

// ---------------- fragment helpers ----------------
// A-fragment (16x32 f16, M rows x K): lane<16 -> M=lane, K in {0..7,16..23};
// lane>=16 -> same M, K in {8..15,24..31}.  src row-major, stride in halves.
__device__ __forceinline__ v16h ldsA(const _Float16* base, int row0, int stride, int k0) {
  const int lane = threadIdx.x & 31;
  const _Float16* p = base + (size_t)(row0 + (lane & 15)) * stride + k0 + ((lane >> 4) << 3);
  v8h lo = *(const v8h*)p;
  v8h hi = *(const v8h*)(p + 16);
  v16h a;
#pragma unroll
  for (int i = 0; i < 8; ++i) { a[i] = lo[i]; a[i + 8] = hi[i]; }
  return a;
}

// A-fragment sourced from f32 global, converted on the fly.
__device__ __forceinline__ v16h ldAg(const float* X, int row0, int stride, int k0) {
  const int lane = threadIdx.x & 31;
  const float* p = X + (size_t)(row0 + (lane & 15)) * stride + k0 + ((lane >> 4) << 3);
  v16h a;
#pragma unroll
  for (int i = 0; i < 8; ++i) { a[i] = (_Float16)p[i]; a[i + 8] = (_Float16)p[16 + i]; }
  return a;
}

// B-fragment (32x16 f16, K rows x N cols): lane<16 -> N=lane, K in {0..7,16..23};
// lane>=16 -> same N, K in {8..15,24..31}.  W row-major [K][...], stride in halves.
__device__ __forceinline__ v16h ldB(const _Float16* W, int k0, int stride, int col0) {
  const int lane = threadIdx.x & 31;
  const _Float16* p = W + (size_t)(k0 + ((lane >> 4) << 3)) * stride + col0 + (lane & 15);
  v16h b;
#pragma unroll
  for (int i = 0; i < 8; ++i) { b[i] = p[(size_t)i * stride]; b[i + 8] = p[(size_t)(16 + i) * stride]; }
  return b;
}

__device__ __forceinline__ v8f wmma_f16(v16h a, v16h b, v8f c) {
  return __builtin_amdgcn_wmma_f32_16x16x32_f16(false, a, false, b, (short)0, c, false, false);
}

__device__ __forceinline__ float sspf(float v) {       // softplus(v) - log(2)
  float s = (v > 20.f) ? v : __logf(1.f + __expf(v));
  return s - LOG2F_;
}

// ---------------- kernels ----------------

// x[b,n,:] = X[:]
__global__ void k_init_x(float* __restrict__ x, const float* __restrict__ X) {
  int idx = blockIdx.x * 256 + threadIdx.x;
  x[idx] = X[idx & (DD - 1)];
}

// Convert / transpose / pad all weights to f16 layouts.
__global__ void k_prep(const float* __restrict__ wW1, const float* __restrict__ wW2,
                       const float* __restrict__ hW, const float* __restrict__ gW,
                       _Float16* __restrict__ wbuf) {
  int tid = blockIdx.x * 256 + threadIdx.x;
  int stride = gridDim.x * 256;
  for (int i = tid; i < 9 * W1_SZ; i += stride) {           // W1^T padded: [f][hpad48]
    int lc = i / W1_SZ, rem = i % W1_SZ, f = rem / HP, h = rem % HP;
    wbuf[W1_OFF + i] = (_Float16)((h < HWID) ? wW1[(size_t)(lc * HWID + h) * FF + f] : 0.f);
  }
  for (int i = tid; i < 9 * W2_SZ; i += stride) {           // W2^T padded: [hpad64][k]
    int lc = i / W2_SZ, rem = i % W2_SZ, h = rem / KK, k = rem % KK;
    wbuf[W2_OFF + i] = (_Float16)((h < HWID) ? wW2[(size_t)(lc * KK + k) * HWID + h] : 0.f);
  }
  for (int i = tid; i < 3 * HWC_SZ; i += stride) {          // hW^T: [d][k]
    int l = i / HWC_SZ, rem = i % HWC_SZ, d = rem / KK, k = rem % KK;
    wbuf[HWC_OFF + i] = (_Float16)hW[(size_t)(l * KK + k) * DD + d];
  }
  for (int i = tid; i < 9 * GW_SZ; i += stride) {           // gW^T: [k][d]
    int lc = i / GW_SZ, rem = i % GW_SZ, k = rem / DD, d = rem % DD;
    wbuf[GW_OFF + i] = (_Float16)gW[(size_t)(lc * DD + d) * KK + k];
  }
}

// h = x @ hW^T + hb    rows = B*N, 128 rows per workgroup, 1 M-tile per wave.
__global__ void k_h(const float* __restrict__ x, const _Float16* __restrict__ hWc,
                    const float* __restrict__ hb, float* __restrict__ h, int l) {
  const int wave = threadIdx.x >> 5, lane = threadIdx.x & 31;
  const int row0 = blockIdx.x * 128 + wave * 16;
  const _Float16* W = hWc + (size_t)l * HWC_SZ;
  v8f acc[4] = {};
#pragma unroll
  for (int ks = 0; ks < 4; ++ks) {
    v16h a = ldAg(x, row0, DD, ks * 32);
#pragma unroll
    for (int nt = 0; nt < 4; ++nt)
      acc[nt] = wmma_f16(a, ldB(W, ks * 32, KK, nt * 16), acc[nt]);
  }
  const int rowH = (lane >> 4) * 8, colL = lane & 15;
#pragma unroll
  for (int nt = 0; nt < 4; ++nt) {
    int col = colL + nt * 16;
    float bias = hb[l * KK + col];
#pragma unroll
    for (int r = 0; r < 8; ++r)
      h[(size_t)(row0 + rowH + r) * KK + col] = acc[nt][r] + bias;
  }
}

// Edge MLP + message aggregation for one (batch, spin-block).
__global__ void k_edge(const float* __restrict__ edges, const float* __restrict__ h,
                       const _Float16* __restrict__ W1c, const _Float16* __restrict__ W2c,
                       const float* __restrict__ wb1, const float* __restrict__ wb2,
                       float* __restrict__ z_same, float* __restrict__ z_anti, int l) {
  __shared__ _Float16 eA[256 * 32];     // edge features f16
  __shared__ _Float16 act[256 * 64];    // ssp activations f16 (H padded to 64)
  __shared__ float    msg[256 * 64];    // MLP output f32

  const int b    = blockIdx.x >> 2;
  const int blk  = blockIdx.x & 3;                 // 0:uu 1:ud 2:du 3:dd
  const int chan = (blk == 1 || blk == 2) ? 1 : 0;
  const int same = (blk == 0 || blk == 3);
  const int rj   = same ? 15 : 16;
  const int iBase = (blk >= 2) ? 16 : 0;
  const int jBase = (blk == 1 || blk == 3) ? 16 : 0;
  const int tid = threadIdx.x;
  const int nrows = 16 * rj;

  // ---- phase 1: gather edges -> LDS (f16), zero act padding cols ----
  {
    _Float16* dst = &eA[tid * 32];
    if (tid < nrows) {
      int il = tid / rj, jj = tid % rj;
      int jg = jBase + jj + ((same && jj >= il) ? 1 : 0);
      const float* src = edges + (((size_t)b * N_TOT + (iBase + il)) * N_TOT + jg) * FF;
#pragma unroll
      for (int f = 0; f < 32; f += 4) {
        float4 v = *(const float4*)(src + f);
        dst[f + 0] = (_Float16)v.x; dst[f + 1] = (_Float16)v.y;
        dst[f + 2] = (_Float16)v.z; dst[f + 3] = (_Float16)v.w;
      }
    } else {
#pragma unroll
      for (int f = 0; f < 32; ++f) dst[f] = (_Float16)0.f;
    }
#pragma unroll
    for (int c = 48; c < 64; ++c) act[tid * 64 + c] = (_Float16)0.f;
  }
  __syncthreads();

  const _Float16* W1 = W1c + (size_t)(l * 3 + chan) * W1_SZ;
  const _Float16* W2 = W2c + (size_t)(l * 3 + chan) * W2_SZ;
  const float* b1 = wb1 + (l * 3 + chan) * HWID;
  const float* b2 = wb2 + (l * 3 + chan) * KK;
  const int wave = tid >> 5, lane = tid & 31;
  const int rowH = (lane >> 4) * 8, colL = lane & 15;

  // ---- phase 2: GEMM1 (F=32 -> Hpad=48) + bias + ssp -> act LDS ----
#pragma unroll
  for (int hf = 0; hf < 2; ++hf) {
    int row0 = (wave + hf * 8) * 16;
    v16h a = ldsA(eA, row0, FF, 0);
#pragma unroll
    for (int nt = 0; nt < 3; ++nt) {
      v8f c = {};
      c = wmma_f16(a, ldB(W1, 0, HP, nt * 16), c);
      int col = colL + nt * 16;
      float bias = (col < HWID) ? b1[col] : 0.f;
#pragma unroll
      for (int r = 0; r < 8; ++r)
        act[(size_t)(row0 + rowH + r) * 64 + col] = (_Float16)sspf(c[r] + bias);
    }
  }
  __syncthreads();

  // ---- phase 3: GEMM2 (Hpad=64 -> K=64) + bias -> msg LDS ----
#pragma unroll
  for (int hf = 0; hf < 2; ++hf) {
    int row0 = (wave + hf * 8) * 16;
    v8f c2[4] = {};
#pragma unroll
    for (int ks = 0; ks < 2; ++ks) {
      v16h a2 = ldsA(act, row0, 64, ks * 32);
#pragma unroll
      for (int nt = 0; nt < 4; ++nt)
        c2[nt] = wmma_f16(a2, ldB(W2, ks * 32, KK, nt * 16), c2[nt]);
    }
#pragma unroll
    for (int nt = 0; nt < 4; ++nt) {
      int col = colL + nt * 16;
      float bias = b2[col];
#pragma unroll
      for (int r = 0; r < 8; ++r)
        msg[(size_t)(row0 + rowH + r) * 64 + col] = c2[nt][r] + bias;
    }
  }
  __syncthreads();

  // ---- phase 4: z[i,k] = sum_j msg[i,j,k] * h[b,jg,k] ----
  {
    int il = tid >> 4;                // receiver 0..15
    int kb = (tid & 15) * 4;          // 4 k's per thread
    float a0 = 0.f, a1 = 0.f, a2 = 0.f, a3 = 0.f;
    for (int jj = 0; jj < rj; ++jj) {
      int jg = jBase + jj + ((same && jj >= il) ? 1 : 0);
      float4 hv = *(const float4*)(h + ((size_t)(b * N_TOT + jg)) * KK + kb);
      const float* mp = &msg[(size_t)(il * rj + jj) * 64 + kb];
      a0 += mp[0] * hv.x; a1 += mp[1] * hv.y; a2 += mp[2] * hv.z; a3 += mp[3] * hv.w;
    }
    float* zdst = (chan == 0 ? z_same : z_anti) + ((size_t)(b * N_TOT + iBase + il)) * KK + kb;
    zdst[0] = a0; zdst[1] = a1; zdst[2] = a2; zdst[3] = a3;
  }
}

// Nuclear channel: w_mlp(l,2,edges_nuc) weighted by Y, summed over m.
__global__ void k_nuc(const float* __restrict__ edges_nuc, const float* __restrict__ Y,
                      const _Float16* __restrict__ W1c, const _Float16* __restrict__ W2c,
                      const float* __restrict__ wb1, const float* __restrict__ wb2,
                      float* __restrict__ z_nuc, int l) {
  __shared__ _Float16 eA[128 * 32];
  __shared__ _Float16 act[128 * 64];
  __shared__ float    msg[128 * 64];
  const int b = blockIdx.x;
  const int tid = threadIdx.x;

  {  // 128 rows (n*4+m), 2 threads per row
    int row = tid >> 1, f0 = (tid & 1) * 16;
    const float* src = edges_nuc + ((size_t)b * 128 + row) * FF + f0;
    _Float16* dst = &eA[row * 32 + f0];
#pragma unroll
    for (int f = 0; f < 16; f += 4) {
      float4 v = *(const float4*)(src + f);
      dst[f + 0] = (_Float16)v.x; dst[f + 1] = (_Float16)v.y;
      dst[f + 2] = (_Float16)v.z; dst[f + 3] = (_Float16)v.w;
    }
    if (tid < 128) {
#pragma unroll
      for (int c = 48; c < 64; ++c) act[tid * 64 + c] = (_Float16)0.f;
    }
  }
  __syncthreads();

  const _Float16* W1 = W1c + (size_t)(l * 3 + 2) * W1_SZ;
  const _Float16* W2 = W2c + (size_t)(l * 3 + 2) * W2_SZ;
  const float* b1 = wb1 + (l * 3 + 2) * HWID;
  const float* b2 = wb2 + (l * 3 + 2) * KK;
  const int wave = tid >> 5, lane = tid & 31;
  const int rowH = (lane >> 4) * 8, colL = lane & 15;
  const int row0 = wave * 16;

  {
    v16h a = ldsA(eA, row0, FF, 0);
#pragma unroll
    for (int nt = 0; nt < 3; ++nt) {
      v8f c = {};
      c = wmma_f16(a, ldB(W1, 0, HP, nt * 16), c);
      int col = colL + nt * 16;
      float bias = (col < HWID) ? b1[col] : 0.f;
#pragma unroll
      for (int r = 0; r < 8; ++r)
        act[(size_t)(row0 + rowH + r) * 64 + col] = (_Float16)sspf(c[r] + bias);
    }
  }
  __syncthreads();
  {
    v8f c2[4] = {};
#pragma unroll
    for (int ks = 0; ks < 2; ++ks) {
      v16h a2 = ldsA(act, row0, 64, ks * 32);
#pragma unroll
      for (int nt = 0; nt < 4; ++nt)
        c2[nt] = wmma_f16(a2, ldB(W2, ks * 32, KK, nt * 16), c2[nt]);
    }
#pragma unroll
    for (int nt = 0; nt < 4; ++nt) {
      int col = colL + nt * 16;
      float bias = b2[col];
#pragma unroll
      for (int r = 0; r < 8; ++r)
        msg[(size_t)(row0 + rowH + r) * 64 + col] = c2[nt][r] + bias;
    }
  }
  __syncthreads();
  {
    int n = tid >> 3, kb = (tid & 7) * 8;
#pragma unroll
    for (int kk = 0; kk < 8; ++kk) {
      int k = kb + kk;
      float a = 0.f;
#pragma unroll
      for (int m = 0; m < MM; ++m) a += msg[(size_t)(n * MM + m) * 64 + k] * Y[m * KK + k];
      z_nuc[((size_t)(b * N_TOT + n)) * KK + k] = a;
    }
  }
}

// x += z_same@gW0^T + z_anti@gW1^T + z_nuc@gW2^T + biases
__global__ void k_update(float* __restrict__ x, const float* __restrict__ z_same,
                         const float* __restrict__ z_anti, const float* __restrict__ z_nuc,
                         const _Float16* __restrict__ gWc, const float* __restrict__ gb, int l) {
  const int wave = threadIdx.x >> 5, lane = threadIdx.x & 31;
  const int row0 = blockIdx.x * 128 + wave * 16;
  v8f acc[8] = {};
#pragma unroll
  for (int c = 0; c < 3; ++c) {
    const float* zp = (c == 0) ? z_same : (c == 1) ? z_anti : z_nuc;
    const _Float16* W = gWc + (size_t)(l * 3 + c) * GW_SZ;
#pragma unroll
    for (int ks = 0; ks < 2; ++ks) {
      v16h a = ldAg(zp, row0, KK, ks * 32);
#pragma unroll
      for (int nt = 0; nt < 8; ++nt)
        acc[nt] = wmma_f16(a, ldB(W, ks * 32, DD, nt * 16), acc[nt]);
    }
  }
  const int rowH = (lane >> 4) * 8, colL = lane & 15;
#pragma unroll
  for (int nt = 0; nt < 8; ++nt) {
    int col = colL + nt * 16;
    float bias = gb[(l * 3 + 0) * DD + col] + gb[(l * 3 + 1) * DD + col] + gb[(l * 3 + 2) * DD + col];
#pragma unroll
    for (int r = 0; r < 8; ++r) {
      size_t idx = (size_t)(row0 + rowH + r) * DD + col;
      x[idx] += acc[nt][r] + bias;
    }
  }
}

extern "C" void kernel_launch(void* const* d_in, const int* in_sizes, int n_in,
                              void* d_out, int out_size, void* d_ws, size_t ws_size,
                              hipStream_t stream) {
  (void)in_sizes; (void)n_in; (void)out_size; (void)ws_size;
  const float* edges_elec = (const float*)d_in[0];
  const float* edges_nuc  = (const float*)d_in[1];
  const float* X   = (const float*)d_in[2];
  const float* Y   = (const float*)d_in[3];
  const float* wW1 = (const float*)d_in[4];
  const float* wb1 = (const float*)d_in[5];
  const float* wW2 = (const float*)d_in[6];
  const float* wb2 = (const float*)d_in[7];
  const float* hW  = (const float*)d_in[8];
  const float* hb  = (const float*)d_in[9];
  const float* gW  = (const float*)d_in[10];
  const float* gb  = (const float*)d_in[11];

  float* x = (float*)d_out;                       // (B, N, D)
  float* h      = (float*)d_ws;                   // (B*N, K)
  float* z_same = h + BUF_ELEMS;
  float* z_anti = z_same + BUF_ELEMS;
  float* z_nuc  = z_anti + BUF_ELEMS;
  _Float16* wbuf = (_Float16*)(z_nuc + BUF_ELEMS);

  k_init_x<<<(BB * N_TOT * DD) / 256, 256, 0, stream>>>(x, X);
  k_prep<<<64, 256, 0, stream>>>(wW1, wW2, hW, gW, wbuf);

  for (int l = 0; l < LL; ++l) {
    k_h<<<(BB * N_TOT) / 128, 256, 0, stream>>>(x, wbuf + HWC_OFF, hb, h, l);
    k_edge<<<BB * 4, 256, 0, stream>>>(edges_elec, h, wbuf + W1_OFF, wbuf + W2_OFF,
                                       wb1, wb2, z_same, z_anti, l);
    k_nuc<<<BB, 256, 0, stream>>>(edges_nuc, Y, wbuf + W1_OFF, wbuf + W2_OFF,
                                  wb1, wb2, z_nuc, l);
    k_update<<<(BB * N_TOT) / 128, 256, 0, stream>>>(x, z_same, z_anti, z_nuc,
                                                     wbuf + GW_OFF, gb, l);
  }
}
